// SelfAttention_79448305041806
// MI455X (gfx1250) — compile-verified, matrix operand order checked by reading, not run on hardware
//
#include <hip/hip_runtime.h>
#include <hip/hip_bf16.h>

typedef __attribute__((ext_vector_type(16))) __bf16 v16bf;
typedef __attribute__((ext_vector_type(8)))  __bf16 v8bf;
typedef __attribute__((ext_vector_type(8)))  float  v8f;

#define DIMC 1024
#define LSEQ 2048
#define NH   16
#define HD   64
#define ASCALE 0.125f   // 64^-0.5

#define WMMA_BF16(a, b, c) \
  __builtin_amdgcn_wmma_f32_16x16x32_bf16(false, (a), false, (b), (short)0, (c), false, false)

// ---- fragment loaders -------------------------------------------------------
// A-matrix 16x32 bf16 layout (CDNA5 ISA 7.12.2): lane m = lane&15,
// element e -> k = (e<8?0:16) + (lane>>4)*8 + (e&7). Same index pattern used
// for the B operand with lane meaning column n. Both map to two contiguous
// 8-element (16B) runs per lane when K is the fast memory axis.

__device__ __forceinline__ v16bf frag_from_f32(const float* __restrict__ base,
                                               int stride, int row, int k0, int lane) {
  const int kk = k0 + ((lane >> 4) << 3);
  const float* p = base + (long)row * stride + kk;
  v16bf a;
#pragma unroll
  for (int i = 0; i < 8; ++i) {
    a[i]     = (__bf16)p[i];
    a[i + 8] = (__bf16)p[i + 16];
  }
  return a;
}

__device__ __forceinline__ v16bf frag_from_bf16(const __bf16* __restrict__ base,
                                                int stride, int row, int k0, int lane) {
  const int kk = k0 + ((lane >> 4) << 3);
  const __bf16* p = base + (long)row * stride + kk;
  v8bf lo = *reinterpret_cast<const v8bf*>(p);
  v8bf hi = *reinterpret_cast<const v8bf*>(p + 16);
  v16bf a;
#pragma unroll
  for (int i = 0; i < 8; ++i) { a[i] = lo[i]; a[i + 8] = hi[i]; }
  return a;
}

// ---- kernel 1: fused QKV projection (z = 0/1/2 -> q/k/v) --------------------
// One wave computes a 16x64 output tile. All four B fragments are built as
// simultaneously-live values before the four WMMAs issue, so the WMMA B-operand
// registers are not immediately overwritten (avoids WMMA->VALU WAR v_nops).
__global__ __launch_bounds__(256) void qkv_proj_k(
    const float* __restrict__ x,
    const float* __restrict__ Wq, const float* __restrict__ bq,
    const float* __restrict__ Wk, const float* __restrict__ bk,
    const float* __restrict__ Wv, const float* __restrict__ bv,
    __bf16* __restrict__ qo, __bf16* __restrict__ ko, __bf16* __restrict__ vto) {
  const int lane = threadIdx.x & 31;
  const int wave = threadIdx.x >> 5;
  const int tile = blockIdx.x * 8 + wave;       // 256 * 16 = 4096 tiles
  const int m0 = (tile >> 4) << 4;              // token tile (16 rows)
  const int n0 = (tile & 15) << 6;              // feature tile (64 cols)
  const int which = blockIdx.z;
  const float* W    = (which == 0) ? Wq : (which == 1) ? Wk : Wv;
  const float* bias = (which == 0) ? bq : (which == 1) ? bk : bv;
  const int ln = lane & 15;

  const v8f zero = {0.f,0.f,0.f,0.f,0.f,0.f,0.f,0.f};
  v8f c0 = zero, c1 = zero, c2 = zero, c3 = zero;

#pragma unroll 2
  for (int k0 = 0; k0 < DIMC; k0 += 32) {
    v16bf a  = frag_from_f32(x, DIMC, m0 + ln, k0, lane);
    v16bf b0 = frag_from_f32(W, DIMC, n0 +  0 + ln, k0, lane);  // B[k][n]=W[n][k]
    v16bf b1 = frag_from_f32(W, DIMC, n0 + 16 + ln, k0, lane);
    v16bf b2 = frag_from_f32(W, DIMC, n0 + 32 + ln, k0, lane);
    v16bf b3 = frag_from_f32(W, DIMC, n0 + 48 + ln, k0, lane);
    c0 = WMMA_BF16(a, b0, c0);
    c1 = WMMA_BF16(a, b1, c1);
    c2 = WMMA_BF16(a, b2, c2);
    c3 = WMMA_BF16(a, b3, c3);
  }
  v8f c[4] = {c0, c1, c2, c3};
  const int hi2 = (lane >> 4) << 3;
#pragma unroll
  for (int t = 0; t < 4; ++t) {
    const int n = n0 + t * 16 + ln;
    const float bb = bias[n];
    const int h = n >> 6, d = n & 63;
#pragma unroll
    for (int r = 0; r < 8; ++r) {
      const int tok = m0 + r + hi2;
      const int b_ = tok >> 11, l = tok & (LSEQ - 1);
      const __bf16 val = (__bf16)(c[t][r] + bb);
      if (which == 2) {                   // V stored transposed: (B,H,D,L)
        vto[(((long)((b_ << 4) + h) * HD + d) * LSEQ) + l] = val;
      } else {                            // Q/K stored (B,H,L,D)
        __bf16* dst = (which == 0) ? qo : ko;
        dst[(((long)((b_ << 4) + h) * LSEQ + l) * HD) + d] = val;
      }
    }
  }
}

// ---- kernel 2: flash attention, one wave per 16-query tile ------------------
__global__ __launch_bounds__(256) void attn_k(
    const __bf16* __restrict__ qb, const __bf16* __restrict__ kb,
    const __bf16* __restrict__ vtb, __bf16* __restrict__ ob) {
  __shared__ __bf16 pstage[8][16 * 32];   // per-wave P staging (D->A relayout)
  const int lane = threadIdx.x & 31;
  const int wave = threadIdx.x >> 5;
  const int w  = blockIdx.x * 8 + wave;   // 32 (b,h) * 128 q-tiles = 4096 waves
  const int bh = w >> 7;
  const int q0 = (w & 127) << 4;
  const int ln = lane & 15;
  const int hi2 = (lane >> 4) << 3;

  const __bf16* qh = qb  + (long)bh * LSEQ * HD;
  const __bf16* kh = kb  + (long)bh * LSEQ * HD;
  const __bf16* vh = vtb + (long)bh * HD * LSEQ;

  // Q fragments live in registers for the whole sweep (16 x 64 = 2 A-frags)
  const v16bf qa0 = frag_from_bf16(qh, HD, q0 + ln, 0,  lane);
  const v16bf qa1 = frag_from_bf16(qh, HD, q0 + ln, 32, lane);

  const v8f zero = {0.f,0.f,0.f,0.f,0.f,0.f,0.f,0.f};
  float mrow[8], lrow[8];
  v8f acc[4];
#pragma unroll
  for (int r = 0; r < 8; ++r) { mrow[r] = -3.0e38f; lrow[r] = 0.f; }
#pragma unroll
  for (int t = 0; t < 4; ++t) acc[t] = zero;

  __bf16* st = pstage[wave];

  for (int j0 = 0; j0 < LSEQ; j0 += 32) {
    // S = (q . k^T) * scale, two 16x16 tiles (keys j0..15, j0+16..31)
    v8f s0 = zero, s1 = zero;
    {
      v16bf b0 = frag_from_bf16(kh, HD, j0 + ln, 0,  lane);
      v16bf b1 = frag_from_bf16(kh, HD, j0 + ln, 32, lane);
      v16bf b2 = frag_from_bf16(kh, HD, j0 + 16 + ln, 0,  lane);
      v16bf b3 = frag_from_bf16(kh, HD, j0 + 16 + ln, 32, lane);
      s0 = WMMA_BF16(qa0, b0, s0);
      s1 = WMMA_BF16(qa0, b2, s1);
      s0 = WMMA_BF16(qa1, b1, s0);
      s1 = WMMA_BF16(qa1, b3, s1);
    }
    // online softmax update; row m = r + 8*(lane>>4) spans a 16-lane group
    float p0[8], p1[8], corr[8];
#pragma unroll
    for (int r = 0; r < 8; ++r) {
      const float a = s0[r] * ASCALE, b = s1[r] * ASCALE;
      float mx = fmaxf(a, b);
#pragma unroll
      for (int msk = 1; msk < 16; msk <<= 1)
        mx = fmaxf(mx, __shfl_xor(mx, msk, 32));
      const float mnew = fmaxf(mrow[r], mx);
      p0[r] = __expf(a - mnew);
      p1[r] = __expf(b - mnew);
      float rs = p0[r] + p1[r];
#pragma unroll
      for (int msk = 1; msk < 16; msk <<= 1)
        rs += __shfl_xor(rs, msk, 32);
      corr[r] = __expf(mrow[r] - mnew);
      lrow[r] = lrow[r] * corr[r] + rs;
      mrow[r] = mnew;
    }
#pragma unroll
    for (int t = 0; t < 4; ++t)
#pragma unroll
      for (int r = 0; r < 8; ++r) acc[t][r] *= corr[r];

    // stage P (16 queries x 32 keys) bf16 row-major in LDS, reload as A-frag.
    // Same-wave DS ops are in-order: no barrier needed, compiler waits DScnt.
#pragma unroll
    for (int r = 0; r < 8; ++r) {
      const int row = r + hi2;
      st[row * 32 + ln]      = (__bf16)p0[r];
      st[row * 32 + 16 + ln] = (__bf16)p1[r];
    }
    const v16bf pa = frag_from_bf16(st, 32, ln, 0, lane);
    {
      v16bf vb0 = frag_from_bf16(vh, LSEQ,  0 + ln, j0, lane);  // B[k][n]=vT[n][k]
      v16bf vb1 = frag_from_bf16(vh, LSEQ, 16 + ln, j0, lane);
      v16bf vb2 = frag_from_bf16(vh, LSEQ, 32 + ln, j0, lane);
      v16bf vb3 = frag_from_bf16(vh, LSEQ, 48 + ln, j0, lane);
      acc[0] = WMMA_BF16(pa, vb0, acc[0]);
      acc[1] = WMMA_BF16(pa, vb1, acc[1]);
      acc[2] = WMMA_BF16(pa, vb2, acc[2]);
      acc[3] = WMMA_BF16(pa, vb3, acc[3]);
    }
  }
  // normalize and emit in (B, L, H*D) bf16 for the output projection
  const int b_ = bh >> 4, h = bh & 15;
#pragma unroll
  for (int r = 0; r < 8; ++r) {
    const float inv = 1.0f / lrow[r];
    const int row = q0 + r + hi2;
#pragma unroll
    for (int t = 0; t < 4; ++t)
      ob[(((long)b_ * LSEQ + row) * DIMC) + h * HD + t * 16 + ln] =
          (__bf16)(acc[t][r] * inv);
  }
}

// ---- kernel 3: output projection (16x64 tile per wave) ----------------------
__global__ __launch_bounds__(256) void oproj_k(
    const __bf16* __restrict__ ai, const float* __restrict__ Wp,
    const float* __restrict__ bp, float* __restrict__ out) {
  const int lane = threadIdx.x & 31;
  const int wave = threadIdx.x >> 5;
  const int tile = blockIdx.x * 8 + wave;       // 256 * 16 tiles
  const int m0 = (tile >> 4) << 4;
  const int n0 = (tile & 15) << 6;
  const int ln = lane & 15;

  const v8f zero = {0.f,0.f,0.f,0.f,0.f,0.f,0.f,0.f};
  v8f c0 = zero, c1 = zero, c2 = zero, c3 = zero;

#pragma unroll 2
  for (int k0 = 0; k0 < DIMC; k0 += 32) {
    v16bf a  = frag_from_bf16(ai, DIMC, m0 + ln, k0, lane);
    v16bf b0 = frag_from_f32(Wp, DIMC, n0 +  0 + ln, k0, lane);
    v16bf b1 = frag_from_f32(Wp, DIMC, n0 + 16 + ln, k0, lane);
    v16bf b2 = frag_from_f32(Wp, DIMC, n0 + 32 + ln, k0, lane);
    v16bf b3 = frag_from_f32(Wp, DIMC, n0 + 48 + ln, k0, lane);
    c0 = WMMA_BF16(a, b0, c0);
    c1 = WMMA_BF16(a, b1, c1);
    c2 = WMMA_BF16(a, b2, c2);
    c3 = WMMA_BF16(a, b3, c3);
  }
  v8f c[4] = {c0, c1, c2, c3};
  const int hi2 = (lane >> 4) << 3;
#pragma unroll
  for (int t = 0; t < 4; ++t) {
    const float bb = bp[n0 + t * 16 + ln];
#pragma unroll
    for (int r = 0; r < 8; ++r)
      out[(long)(m0 + r + hi2) * DIMC + n0 + t * 16 + ln] = c[t][r] + bb;
  }
}

// ---- launch -----------------------------------------------------------------
extern "C" void kernel_launch(void* const* d_in, const int* in_sizes, int n_in,
                              void* d_out, int out_size, void* d_ws, size_t ws_size,
                              hipStream_t stream) {
  const float* x  = (const float*)d_in[0];
  const float* Wq = (const float*)d_in[1];
  const float* bq = (const float*)d_in[2];
  const float* Wk = (const float*)d_in[3];
  const float* bk = (const float*)d_in[4];
  const float* Wv = (const float*)d_in[5];
  const float* bv = (const float*)d_in[6];
  const float* Wp = (const float*)d_in[7];
  const float* bp = (const float*)d_in[8];

  char* ws = (char*)d_ws;
  __bf16* qb  = (__bf16*)(ws);                          //  8 MB (B,H,L,D)
  __bf16* kb  = (__bf16*)(ws + ((size_t)8  << 20));     //  8 MB (B,H,L,D)
  __bf16* vtb = (__bf16*)(ws + ((size_t)16 << 20));     //  8 MB (B,H,D,L)
  __bf16* ob  = (__bf16*)(ws + ((size_t)24 << 20));     //  8 MB (B,L,C)

  qkv_proj_k<<<dim3(512, 1, 3), 256, 0, stream>>>(x, Wq, bq, Wk, bk, Wv, bv,
                                                  qb, kb, vtb);
  attn_k<<<dim3(512), 256, 0, stream>>>(qb, kb, vtb, ob);
  oproj_k<<<dim3(512), 256, 0, stream>>>(ob, Wp, bp, (float*)d_out);
}
